// Head_360777253116
// MI455X (gfx1250) — matrix operand order repaired and checked by reading, not compiled, not run on hardware
//
#include <hip/hip_runtime.h>

typedef __attribute__((ext_vector_type(16))) _Float16 v16h;
typedef __attribute__((ext_vector_type(8)))  _Float16 v8h;
typedef __attribute__((ext_vector_type(8)))  float    v8f;
typedef __attribute__((ext_vector_type(8)))  float    f32x8;

#define B_DIM 4
#define T_DIM 4096
#define C_DIM 512
#define H_DIM 64

static __device__ __forceinline__ v8f wmma_f16f32(v16h a, v16h b, v8f c) {
  // D = A(16x32 f16) * B(32x16 f16) + C(16x16 f32)
  return __builtin_amdgcn_wmma_f32_16x16x32_f16(false, a, false, b,
                                                (short)0, c, false, false);
}

// ---------------------------------------------------------------------------
// Kernel 0: one-shot W -> f16 B-operand staging.
// Layout: wstage[(((mat*16 + kk32)*4 + t)*32 + lane)*16 + i]
//   where the 16 halves per lane are exactly the B-operand register contents
//   for WMMA tile (kk = kk32*32, n-tile t):  k = kk + 8*hs + i + 8*(i>=8), col = t*16+l15.
// 192 KB total -> stays resident in L2 for the projection kernel.
// ---------------------------------------------------------------------------
__global__ void __launch_bounds__(256)
wstage_kernel(const float* __restrict__ Wq,
              const float* __restrict__ Wk,
              const float* __restrict__ Wv,
              _Float16* __restrict__ wstage)
{
  const int idx = blockIdx.x * 256 + threadIdx.x;   // 3*16*4*32*16 = 98304
  if (idx >= 3 * 16 * 4 * 512) return;
  const int i    = idx & 15;
  const int lane = (idx >> 4) & 31;
  const int t    = (idx >> 9) & 3;
  const int kk32 = (idx >> 11) & 15;
  const int mat  = idx >> 15;
  const float* W = (mat == 0) ? Wq : (mat == 1) ? Wk : Wv;
  const int hs = lane >> 4, l15 = lane & 15;
  const int k   = kk32 * 32 + (i < 8 ? 8 * hs + i : 16 + 8 * hs + (i - 8));
  const int col = t * 16 + l15;
  wstage[idx] = (_Float16)W[(size_t)k * H_DIM + col];
}

// ---------------------------------------------------------------------------
// Kernel 1: QKV projection.  One wave = one 16-row tile of one batch.
// A operand: x rows, f32 -> f16 via two 32B vector loads + pack converts.
// B operand: single 32B v16h load from the pre-swizzled wstage.
// q is pre-scaled by 1/sqrt(C); v stored transposed [H][T].
// ---------------------------------------------------------------------------
__global__ void __launch_bounds__(128)
qkv_proj_kernel(const float* __restrict__ x,
                const _Float16* __restrict__ wstage,
                _Float16* __restrict__ qbuf,   // [B][T][H] f16 (pre-scaled)
                _Float16* __restrict__ kbuf,   // [B][T][H] f16
                _Float16* __restrict__ vT)     // [B][H][T] f16
{
  const int lane = threadIdx.x & 31;
  const int wave = threadIdx.x >> 5;
  const int tile = blockIdx.x * 4 + wave;      // 1024 tiles total
  const int b    = tile >> 8;                  // T/16 = 256 tiles per batch
  const int m0   = (tile & 255) << 4;
  const int l15  = lane & 15;
  const int hs   = lane >> 4;

  const float* xb = x + (size_t)b * T_DIM * C_DIM;

  v8f acc[12];                                 // [q0..3, k0..3, v0..3]
  #pragma unroll
  for (int i = 0; i < 12; ++i)
    #pragma unroll
    for (int r = 0; r < 8; ++r) acc[i][r] = 0.0f;

  for (int kk32 = 0; kk32 < C_DIM / 32; ++kk32) {
    // ---- A operand: rows m0+l15 of x; K-pattern = two contiguous 8-float runs
    v16h a;
    {
      const float* xrow = xb + (size_t)(m0 + l15) * C_DIM + kk32 * 32;
      const f32x8 lo = *(const f32x8*)(xrow + 8 * hs);        // K = 8*hs .. +7
      const f32x8 hi = *(const f32x8*)(xrow + 16 + 8 * hs);   // K = 16+8*hs .. +7
      #pragma unroll
      for (int i = 0; i < 8; ++i) {
        a[i]     = (_Float16)lo[i];
        a[i + 8] = (_Float16)hi[i];
      }
    }
    // ---- 12 WMMAs, each B operand is one 32B load of staged f16 ----
    const _Float16* wrow = wstage + (((size_t)kk32 * 4) * 32 + lane) * 16;
    #pragma unroll
    for (int mat = 0; mat < 3; ++mat) {
      #pragma unroll
      for (int t = 0; t < 4; ++t) {
        const v16h bb = *(const v16h*)(wrow +
            ((size_t)mat * 16 * 4 + t) * 512);   // (mat,kk32,t) tile stride
        acc[mat * 4 + t] = wmma_f16f32(a, bb, acc[mat * 4 + t]);
      }
    }
  }

  const float qscale = 0.04419417382415922f;   // 1/sqrt(512)
  #pragma unroll
  for (int t = 0; t < 4; ++t) {
    #pragma unroll
    for (int r = 0; r < 8; ++r) {
      const int m = m0 + r + 8 * hs;           // C-layout row
      const int h = t * 16 + l15;              // C-layout col
      qbuf[((size_t)b * T_DIM + m) * H_DIM + h] = (_Float16)(acc[t][r] * qscale);
      kbuf[((size_t)b * T_DIM + m) * H_DIM + h] = (_Float16)acc[4 + t][r];
      vT [((size_t)b * H_DIM + h) * T_DIM + m] = (_Float16)acc[8 + t][r];
    }
  }
}

// ---------------------------------------------------------------------------
// Kernel 2: causal flash attention.  One wave = one 16-query tile; 32 keys
// per step (4 score WMMAs over the K=64 contraction, online softmax,
// 4 P@V WMMAs).  P's C-layout -> A-layout transpose goes through per-wave LDS
// (same-wave DS ops are in-order; no barrier, so divergent trip counts are OK).
// ---------------------------------------------------------------------------
__global__ void __launch_bounds__(128)
flash_attn_kernel(const _Float16* __restrict__ qbuf,
                  const _Float16* __restrict__ kbuf,
                  const _Float16* __restrict__ vT,
                  float* __restrict__ out)
{
  __shared__ __align__(32) _Float16 pstage[4][16][32];   // per-wave 1KB

  const int lane = threadIdx.x & 31;
  const int wave = threadIdx.x >> 5;
  const int tile = blockIdx.x * 4 + wave;
  const int b    = tile >> 8;
  const int m0   = (tile & 255) << 4;
  const int l15  = lane & 15;
  const int hs   = lane >> 4;

  const _Float16* qb = qbuf + (size_t)b * T_DIM * H_DIM;
  const _Float16* kb = kbuf + (size_t)b * T_DIM * H_DIM;
  const _Float16* vb = vT   + (size_t)b * H_DIM * T_DIM;

  // Resident Q tile: two A operands; each = two contiguous 16B half-runs
  v16h qa[2];
  {
    const _Float16* qrow = qb + (size_t)(m0 + l15) * H_DIM;
    #pragma unroll
    for (int half = 0; half < 2; ++half) {
      const v8h lo = *(const v8h*)(qrow + half * 32 + 16 * hs);
      const v8h hi = *(const v8h*)(qrow + half * 32 + 16 * hs + 8);
      #pragma unroll
      for (int i = 0; i < 8; ++i) {
        qa[half][i]     = lo[i];
        qa[half][i + 8] = hi[i];
      }
    }
  }

  v8f o[4];
  float M[8], L[8];
  #pragma unroll
  for (int t = 0; t < 4; ++t)
    #pragma unroll
    for (int r = 0; r < 8; ++r) o[t][r] = 0.0f;
  #pragma unroll
  for (int r = 0; r < 8; ++r) { M[r] = -3.0e38f; L[r] = 0.0f; }

  // causal bound: keys 0 .. m0+15; j multiple of 32 => max key index 4095
  for (int j = 0; j <= m0 + 15; j += 32) {
    // ---- scores: S[16q x 32k], f32 accum, already scaled (q pre-scaled) ----
    v8f s0, s1;
    #pragma unroll
    for (int r = 0; r < 8; ++r) { s0[r] = 0.0f; s1[r] = 0.0f; }
    #pragma unroll
    for (int half = 0; half < 2; ++half) {
      const v16h bk0 = *(const v16h*)(kb + (size_t)(j + l15)      * H_DIM + half * 32 + 16 * hs);
      const v16h bk1 = *(const v16h*)(kb + (size_t)(j + 16 + l15) * H_DIM + half * 32 + 16 * hs);
      s0 = wmma_f16f32(qa[half], bk0, s0);
      s1 = wmma_f16f32(qa[half], bk1, s1);
    }

    // ---- causal mask + online softmax (row = 16-lane group, C-layout) ----
    float alpha[8];
    #pragma unroll
    for (int r = 0; r < 8; ++r) {
      const int qi = m0 + r + 8 * hs;
      float a0 = (j + l15      > qi) ? -1.0e30f : s0[r];
      float a1 = (j + 16 + l15 > qi) ? -1.0e30f : s1[r];
      float mx = fmaxf(a0, a1);
      mx = fmaxf(mx, __shfl_xor(mx, 1));
      mx = fmaxf(mx, __shfl_xor(mx, 2));
      mx = fmaxf(mx, __shfl_xor(mx, 4));
      mx = fmaxf(mx, __shfl_xor(mx, 8));
      const float Mn = fmaxf(M[r], mx);
      const float al = __expf(M[r] - Mn);
      const float p0 = __expf(a0 - Mn);
      const float p1 = __expf(a1 - Mn);
      float rs = p0 + p1;
      rs += __shfl_xor(rs, 1);
      rs += __shfl_xor(rs, 2);
      rs += __shfl_xor(rs, 4);
      rs += __shfl_xor(rs, 8);
      L[r] = al * L[r] + rs;
      M[r] = Mn;
      alpha[r] = al;
      // stage P (f16) to LDS at [m][key-j]  (C-layout coords)
      const int m = r + 8 * hs;
      pstage[wave][m][l15]      = (_Float16)p0;
      pstage[wave][m][16 + l15] = (_Float16)p1;
    }

    // rescale running output
    #pragma unroll
    for (int t = 0; t < 4; ++t)
      #pragma unroll
      for (int r = 0; r < 8; ++r) o[t][r] *= alpha[r];

    // ---- P@V: read P back in A-layout (two 16B LDS vector loads) ----
    v16h pa;
    {
      const v8h lo = *(const v8h*)&pstage[wave][l15][16 * hs];
      const v8h hi = *(const v8h*)&pstage[wave][l15][16 * hs + 8];
      #pragma unroll
      for (int i = 0; i < 8; ++i) { pa[i] = lo[i]; pa[i + 8] = hi[i]; }
    }
    #pragma unroll
    for (int t = 0; t < 4; ++t) {
      const int h = t * 16 + l15;
      const v16h bv = *(const v16h*)(vb + (size_t)h * T_DIM + j + 16 * hs);
      o[t] = wmma_f16f32(pa, bv, o[t]);
    }
  }

  // ---- finalize: divide by softmax denominator, store f32 ----
  float* ob = out + (size_t)b * T_DIM * H_DIM;
  #pragma unroll
  for (int t = 0; t < 4; ++t)
    #pragma unroll
    for (int r = 0; r < 8; ++r) {
      const int m = m0 + r + 8 * hs;
      const int h = t * 16 + l15;
      ob[(size_t)m * H_DIM + h] = o[t][r] / L[r];
    }
}

// ---------------------------------------------------------------------------
extern "C" void kernel_launch(void* const* d_in, const int* in_sizes, int n_in,
                              void* d_out, int out_size, void* d_ws, size_t ws_size,
                              hipStream_t stream) {
  (void)in_sizes; (void)n_in; (void)out_size; (void)ws_size;
  const float* x  = (const float*)d_in[0];
  const float* Wq = (const float*)d_in[1];
  const float* Wk = (const float*)d_in[2];
  const float* Wv = (const float*)d_in[3];
  float* out = (float*)d_out;

  // workspace: q (2MB) | k (2MB) | vT (2MB) | wstage (192KB), all f16
  _Float16* qbuf   = (_Float16*)d_ws;
  _Float16* kbuf   = qbuf + (size_t)B_DIM * T_DIM * H_DIM;
  _Float16* vTb    = kbuf + (size_t)B_DIM * T_DIM * H_DIM;
  _Float16* wstage = vTb + (size_t)B_DIM * T_DIM * H_DIM;

  const int tiles  = B_DIM * (T_DIM / 16);   // 1024
  const int blocks = tiles / 4;              // 4 waves per block

  wstage_kernel<<<(3 * 16 * 4 * 512 + 255) / 256, 256, 0, stream>>>(Wq, Wk, Wv, wstage);
  qkv_proj_kernel<<<blocks, 128, 0, stream>>>(x, wstage, qbuf, kbuf, vTb);
  flash_attn_kernel<<<blocks, 128, 0, stream>>>(qbuf, kbuf, vTb, out);
}
